// LinearAttentionBridge_75548474736741
// MI455X (gfx1250) — matrix operand order repaired and checked by reading, not compiled, not run on hardware
//
#include <hip/hip_runtime.h>

// ---------------------------------------------------------------------------
// LinearAttentionBridge for MI455X (gfx1250, wave32, WMMA)
// All matrix stages on v_wmma_f32_16x16x32_bf16; every fragment is a
// contiguous b128 load. GEMM staging uses GLOBAL_LOAD_ASYNC_TO_LDS_B128
// (ASYNCcnt) when the toolchain exposes it, else global->VGPR->LDS fallback.
// ws requirement: ~201 MB (bf16 copies + per-head intermediates).
// ---------------------------------------------------------------------------

typedef __attribute__((ext_vector_type(16))) __bf16 v16bf;
typedef __attribute__((ext_vector_type(8)))  float  v8f;
typedef __attribute__((ext_vector_type(4)))  int    v4i;

#define BATCH  4
#define SEQ    4096
#define DMODEL 1024
#define HEADS  16
#define HDIM   64
#define MTOT   (BATCH * SEQ)   // 16384
#define KVSPLIT 8

#if __has_builtin(__builtin_amdgcn_global_load_async_to_lds_b128) && \
    __has_builtin(__builtin_amdgcn_s_wait_asynccnt)
#define HAVE_ASYNC_LDS 1
#define AS1 __attribute__((address_space(1)))
#define AS3 __attribute__((address_space(3)))
#else
#define HAVE_ASYNC_LDS 0
#endif

__device__ __forceinline__ unsigned int pack_bf16x2(float a, float b) {
  union { __bf16 h[2]; unsigned int u; } x;
  x.h[0] = (__bf16)a; x.h[1] = (__bf16)b;       // RNE
  return x.u;
}
__device__ __forceinline__ unsigned short f2bf(float f) {
  union { __bf16 h; unsigned short u; } x; x.h = (__bf16)f; return x.u;
}
__device__ __forceinline__ float bf2f(unsigned short s) {
  union { unsigned int u; float f; } x; x.u = ((unsigned int)s) << 16;
  return x.f;
}

union Frag16 { uint4 q[2]; v16bf v; unsigned short s[16]; };

// ---------------------------------------------------------------------------
// One-time f32 -> bf16 conversion (float4 in, 4xbf16 out).
// ---------------------------------------------------------------------------
__global__ __launch_bounds__(256)
void cvt_bf16(const float* __restrict__ src, unsigned short* __restrict__ dst,
              int n4) {
  int i = blockIdx.x * 256 + threadIdx.x;
  if (i < n4) {
    float4 v = ((const float4*)src)[i];
    uint2 p;
    p.x = pack_bf16x2(v.x, v.y);
    p.y = pack_bf16x2(v.z, v.w);
    ((uint2*)dst)[i] = p;
  }
}

// ---------------------------------------------------------------------------
// Tiled GEMM: C[M,N] = A[M,K](bf16) * W[N,K](bf16)^T, f32 accumulate.
// MODE 0: q-proj -> elu(x*0.125)+1, bf16, head-major      [B,H,SEQ,HDIM]
// MODE 1: k-proj -> elu(x)+1,       bf16, head-TRANSPOSED [B,H,HDIM,SEQ]
// MODE 2: v-proj -> raw,            bf16, head-TRANSPOSED [B,H,HDIM,SEQ]
// MODE 3: o-proj -> raw f32, [B,SEQ,DMODEL] (final output)
// 256 threads = 8 waves (4x2), block tile 128x128, K-step 32, LDS dbl-buffer.
// ---------------------------------------------------------------------------
template<int MODE>
__global__ __launch_bounds__(256)
void gemm_bf16(const unsigned short* __restrict__ A,
               const unsigned short* __restrict__ Wb,
               void* __restrict__ Out) {
  constexpr int K = DMODEL, N = DMODEL;
  constexpr int LDT = 40;                       // padded LDS stride (bf16)
  __shared__ unsigned short As[2][128 * LDT];
  __shared__ unsigned short Bs[2][128 * LDT];

  const int t    = threadIdx.x;
  const int mb   = blockIdx.y, nb = blockIdx.x;
  const int lane = t & 31,  wid = t >> 5;
  const int wm   = wid & 3, wn  = wid >> 2;
  const int r16  = lane & 15, hi = lane >> 4;

  auto loadTiles = [&](int kk, int buf) {
    #pragma unroll
    for (int i = 0; i < 2; ++i) {
      int idx = i * 256 + t;                    // 512 x uint4 per matrix
      int row = idx >> 2, c = idx & 3;
      const unsigned short* gA = A  + (size_t)(mb * 128 + row) * K + kk + c * 8;
      const unsigned short* gB = Wb + (size_t)(nb * 128 + row) * K + kk + c * 8;
      unsigned short* lA = &As[buf][row * LDT + c * 8];
      unsigned short* lB = &Bs[buf][row * LDT + c * 8];
#if HAVE_ASYNC_LDS
      __builtin_amdgcn_global_load_async_to_lds_b128(
          (AS1 v4i*)gA, (AS3 v4i*)lA, 0, 0);
      __builtin_amdgcn_global_load_async_to_lds_b128(
          (AS1 v4i*)gB, (AS3 v4i*)lB, 0, 0);
#else
      *(uint4*)lA = *(const uint4*)gA;
      *(uint4*)lB = *(const uint4*)gB;
#endif
    }
  };
  auto waitTiles = [&]() {
#if HAVE_ASYNC_LDS
    __builtin_amdgcn_s_wait_asynccnt(0);
#endif
    __syncthreads();
  };

  v8f acc[2][4] = {};
  loadTiles(0, 0);
  waitTiles();

  for (int kk = 0; kk < K; kk += 32) {
    const int buf = (kk >> 5) & 1;
    if (kk + 32 < K) loadTiles(kk + 32, buf ^ 1);   // in flight during compute

    Frag16 fa[2], fb[4];
    #pragma unroll
    for (int mt = 0; mt < 2; ++mt) {
      int base = (wm * 32 + mt * 16 + r16) * LDT + hi * 8;  // K=hi*8+i / 16+hi*8+i
      fa[mt].q[0] = *(const uint4*)&As[buf][base];
      fa[mt].q[1] = *(const uint4*)&As[buf][base + 16];
    }
    #pragma unroll
    for (int nt = 0; nt < 4; ++nt) {
      int base = (wn * 64 + nt * 16 + r16) * LDT + hi * 8;  // K=hi*8+i / 16+hi*8+i
      fb[nt].q[0] = *(const uint4*)&Bs[buf][base];
      fb[nt].q[1] = *(const uint4*)&Bs[buf][base + 16];
    }
    #pragma unroll
    for (int mt = 0; mt < 2; ++mt)
      #pragma unroll
      for (int nt = 0; nt < 4; ++nt)
        acc[mt][nt] = __builtin_amdgcn_wmma_f32_16x16x32_bf16(
            false, fa[mt].v, false, fb[nt].v, (short)0, acc[mt][nt], false, false);
    waitTiles();
  }

  // epilogue: D element r -> row hi*8+r, col r16
  #pragma unroll
  for (int mt = 0; mt < 2; ++mt)
    #pragma unroll
    for (int nt = 0; nt < 4; ++nt)
      #pragma unroll
      for (int r = 0; r < 8; ++r) {
        int m = mb * 128 + wm * 32 + mt * 16 + hi * 8 + r;
        int n = nb * 128 + wn * 64 + nt * 16 + r16;
        float val = acc[mt][nt][r];
        if constexpr (MODE == 3) {
          ((float*)Out)[(size_t)m * N + n] = val;
        } else {
          if constexpr (MODE == 0) val *= 0.125f;              // HEAD_DIM^-0.5
          if constexpr (MODE == 0 || MODE == 1)
            val = (val > 0.f) ? (val + 1.f) : __expf(val);     // elu(x)+1
          int b = m >> 12, l = m & (SEQ - 1);
          int h = n >> 6,  d = n & (HDIM - 1);
          size_t idx;
          if constexpr (MODE == 0)          // head-major [B,H,SEQ,HDIM]
            idx = (((size_t)b * HEADS + h) * SEQ + l) * HDIM + d;
          else                               // head-transposed [B,H,HDIM,SEQ]
            idx = (((size_t)b * HEADS + h) * HDIM + d) * SEQ + l;
          ((unsigned short*)Out)[idx] = f2bf(val);
        }
      }
}

// ---------------------------------------------------------------------------
// kvT[head][c][d] += (k_feat^T @ v)[d][c] over an L-slice.
// kfT/vT are [B,H,HDIM,SEQ] so A and B fragments are contiguous b128 global
// loads (no LDS, no barriers). All 10 loads clause before the 4 WMMAs;
// unroll 2 lets next-iteration loads overlap current WMMAs.
// ---------------------------------------------------------------------------
__global__ __launch_bounds__(128)
void kv_accum(const unsigned short* __restrict__ kfT,
              const unsigned short* __restrict__ vT,
              float* __restrict__ kvT) {
  const int t = threadIdx.x;
  const int head = blockIdx.x / KVSPLIT, slice = blockIdx.x % KVSPLIT;
  const int lane = t & 31, w = t >> 5;
  const int r16 = lane & 15, hi = lane >> 4;

  const unsigned short* ka = kfT + ((size_t)head * HDIM + w * 16 + r16) * SEQ;
  const unsigned short* vbase = vT + (size_t)head * HDIM * SEQ;

  v8f acc[4] = {};
  const int l0 = slice * (SEQ / KVSPLIT);
  #pragma unroll 2
  for (int lb = l0; lb < l0 + SEQ / KVSPLIT; lb += 32) {
    Frag16 fa, fb[4];
    fa.q[0] = *(const uint4*)(ka + lb + hi * 8);        // K = hi*8 + i
    fa.q[1] = *(const uint4*)(ka + lb + 16 + hi * 8);   // K = 16 + hi*8 + i
    #pragma unroll
    for (int nt = 0; nt < 4; ++nt) {
      const unsigned short* vrow =
          vbase + (size_t)(nt * 16 + r16) * SEQ + lb + hi * 8;   // K = hi*8+i / 16+hi*8+i
      fb[nt].q[0] = *(const uint4*)(vrow);
      fb[nt].q[1] = *(const uint4*)(vrow + 16);
    }
    #pragma unroll
    for (int nt = 0; nt < 4; ++nt)
      acc[nt] = __builtin_amdgcn_wmma_f32_16x16x32_bf16(
          false, fa.v, false, fb[nt].v, (short)0, acc[nt], false, false);
  }
  #pragma unroll
  for (int nt = 0; nt < 4; ++nt)
    #pragma unroll
    for (int r = 0; r < 8; ++r) {
      int d = w * 16 + hi * 8 + r;          // D row  = kv row index d
      int c = nt * 16 + r16;                // D col  = kv col index c
      atomicAdd(&kvT[(size_t)head * HDIM * HDIM + (size_t)c * HDIM + d],
                acc[nt][r]);
    }
}

// k_sum[head][d] = sum_l kfT[head][d][l]  (contiguous scan, uint4 unpack)
__global__ __launch_bounds__(64)
void ksum_kernel(const unsigned short* __restrict__ kfT,
                 float* __restrict__ ksum) {
  const int head = blockIdx.x, d = threadIdx.x;
  const unsigned short* p = kfT + ((size_t)head * HDIM + d) * SEQ;
  float s = 0.f;
  for (int l = 0; l < SEQ; l += 8) {
    __builtin_prefetch(p + l + 256, 0, 1);          // global_prefetch_b8
    union { uint4 q; unsigned short s16[8]; } u;
    u.q = *(const uint4*)(p + l);
    #pragma unroll
    for (int j = 0; j < 8; ++j) s += bf2f(u.s16[j]);
  }
  ksum[head * HDIM + d] = s;
}

__global__ void zero_f32(float* p, int n) {
  int i = blockIdx.x * 256 + threadIdx.x;
  if (i < n) p[i] = 0.f;
}

// ---------------------------------------------------------------------------
// out = q_feat @ [kv | k_sum]: KV2^T staged row-major in LDS ([c][d]) so
// every B fragment is two contiguous ds_load_b128. nt=4 col 64 = denominator.
// Normalize and store bf16 attn [B,SEQ,DMODEL] for the final o-projection.
// ---------------------------------------------------------------------------
__global__ __launch_bounds__(256)
void attn_norm(const unsigned short* __restrict__ qf,
               const float* __restrict__ kvT, const float* __restrict__ ks,
               unsigned short* __restrict__ attn) {
  constexpr int LDT = 72;                     // 144B stride -> conflict-free
  __shared__ unsigned short KV2T[80 * LDT];   // rows c=0..63 kv, 64 ksum
  __shared__ float denomS[8][16];
  const int t = threadIdx.x;
  const int head = blockIdx.y;
  const int b = head >> 4, h = head & 15;
  const int lblk = blockIdx.x * 128;
  const int lane = t & 31, w = t >> 5;
  const int r16 = lane & 15, hi = lane >> 4;

  #pragma unroll
  for (int i = 0; i < 16; ++i) {
    int idx = i * 256 + t;                    // 4096 entries, coalesced
    int c = idx >> 6, d = idx & 63;
    KV2T[c * LDT + d] = f2bf(kvT[(size_t)head * HDIM * HDIM + idx]);
  }
  if (t < HDIM) KV2T[64 * LDT + t] = f2bf(ks[head * HDIM + t]);
  for (int z = t; z < 15 * 64; z += 256) {    // hygiene: rows 65..79 = 0
    int c = 65 + (z >> 6), d = z & 63;
    KV2T[c * LDT + d] = 0;
  }
  __syncthreads();

  const unsigned short* qrow =
      qf + ((size_t)head * SEQ + lblk + w * 16 + r16) * HDIM;

  v8f acc[5] = {};
  #pragma unroll
  for (int kstep = 0; kstep < 2; ++kstep) {
    Frag16 fa, fb[5];                         // q rows contiguous in d
    fa.q[0] = *(const uint4*)(qrow + kstep * 32 + hi * 8);
    fa.q[1] = *(const uint4*)(qrow + kstep * 32 + 16 + hi * 8);
    #pragma unroll
    for (int nt = 0; nt < 5; ++nt) {
      int base = (nt * 16 + r16) * LDT + kstep * 32 + hi * 8;
      fb[nt].q[0] = *(const uint4*)&KV2T[base];
      fb[nt].q[1] = *(const uint4*)&KV2T[base + 16];
    }
    #pragma unroll
    for (int nt = 0; nt < 5; ++nt)
      acc[nt] = __builtin_amdgcn_wmma_f32_16x16x32_bf16(
          false, fa.v, false, fb[nt].v, (short)0, acc[nt], false, false);
  }

  if (r16 == 0) {                             // lanes 0/16 hold column 64
    #pragma unroll
    for (int r = 0; r < 8; ++r)
      denomS[w][hi * 8 + r] = fmaxf(acc[4][r], 1e-6f);
  }
  __syncthreads();

  #pragma unroll
  for (int nt = 0; nt < 4; ++nt)
    #pragma unroll
    for (int r = 0; r < 8; ++r) {
      int row = hi * 8 + r;
      float val = acc[nt][r] / denomS[w][row];
      int l = lblk + w * 16 + row;
      attn[((size_t)b * SEQ + l) * DMODEL + h * 64 + nt * 16 + r16] = f2bf(val);
    }
}

// ---------------------------------------------------------------------------
extern "C" void kernel_launch(void* const* d_in, const int* in_sizes, int n_in,
                              void* d_out, int out_size, void* d_ws, size_t ws_size,
                              hipStream_t stream) {
  const float* dec = (const float*)d_in[0];
  const float* enc = (const float*)d_in[1];
  const float* wq  = (const float*)d_in[2];
  const float* wk  = (const float*)d_in[3];
  const float* wv  = (const float*)d_in[4];
  const float* wo  = (const float*)d_in[5];
  float* out = (float*)d_out;

  char* ws = (char*)d_ws;
  const size_t actElems  = (size_t)MTOT * DMODEL;                 // 16M
  const size_t wElems    = (size_t)DMODEL * DMODEL;               // 1M
  const size_t headElems = (size_t)BATCH * HEADS * SEQ * HDIM;    // 16M

  unsigned short* decB = (unsigned short*)ws; ws += actElems * 2;   // 32 MB
  unsigned short* encB = (unsigned short*)ws; ws += actElems * 2;   // 32 MB
  unsigned short* wqB  = (unsigned short*)ws; ws += wElems * 2;     // 2 MB
  unsigned short* wkB  = (unsigned short*)ws; ws += wElems * 2;
  unsigned short* wvB  = (unsigned short*)ws; ws += wElems * 2;
  unsigned short* woB  = (unsigned short*)ws; ws += wElems * 2;
  unsigned short* qf   = (unsigned short*)ws; ws += headElems * 2;  // 32 MB
  unsigned short* kfT  = (unsigned short*)ws; ws += headElems * 2;  // 32 MB
  unsigned short* vT   = (unsigned short*)ws; ws += headElems * 2;  // 32 MB
  unsigned short* attn = (unsigned short*)ws; ws += headElems * 2;  // 32 MB
  float* kvbuf = (float*)ws; ws += (size_t)BATCH * HEADS * HDIM * HDIM * 4;
  float* ksbuf = (float*)ws;

  // one-time bf16 conversions
  {
    int n4 = (int)(actElems / 4);
    cvt_bf16<<<(n4 + 255) / 256, 256, 0, stream>>>(dec, decB, n4);
    cvt_bf16<<<(n4 + 255) / 256, 256, 0, stream>>>(enc, encB, n4);
    int w4 = (int)(wElems / 4);
    cvt_bf16<<<(w4 + 255) / 256, 256, 0, stream>>>(wq, wqB, w4);
    cvt_bf16<<<(w4 + 255) / 256, 256, 0, stream>>>(wk, wkB, w4);
    cvt_bf16<<<(w4 + 255) / 256, 256, 0, stream>>>(wv, wvB, w4);
    cvt_bf16<<<(w4 + 255) / 256, 256, 0, stream>>>(wo, woB, w4);
  }

  dim3 gblk(256), ggrid(DMODEL / 128, MTOT / 128);   // 8 x 128 blocks

  gemm_bf16<0><<<ggrid, gblk, 0, stream>>>(decB, wqB, qf);
  gemm_bf16<1><<<ggrid, gblk, 0, stream>>>(encB, wkB, kfT);
  gemm_bf16<2><<<ggrid, gblk, 0, stream>>>(encB, wvB, vT);

  const int nkv = BATCH * HEADS * HDIM * HDIM;
  zero_f32<<<(nkv + 255) / 256, 256, 0, stream>>>(kvbuf, nkv);
  kv_accum<<<BATCH * HEADS * KVSPLIT, 128, 0, stream>>>(kfT, vT, kvbuf);
  ksum_kernel<<<BATCH * HEADS, 64, 0, stream>>>(kfT, ksbuf);

  attn_norm<<<dim3(SEQ / 128, BATCH * HEADS), 256, 0, stream>>>(qf, kvbuf, ksbuf, attn);

  gemm_bf16<3><<<ggrid, gblk, 0, stream>>>(attn, woB, out);
}
